// EmaResidualVectorQuantizer_62345745268868
// MI455X (gfx1250) — matrix operand order repaired and checked by reading, not compiled, not run on hardware
//
#include <hip/hip_runtime.h>

typedef __attribute__((ext_vector_type(16))) __bf16 v16bf;
typedef __attribute__((ext_vector_type(8)))  __bf16 v8bf;
typedef __attribute__((ext_vector_type(8)))  float  v8f;

#define N_TOK   65536   // 64 * 1024 tokens
#define DIM     256
#define KCODES  1024
#define MLEV    4
#define ROWS_WG 128
#define THREADS 256
#define WAVES   8
#define BETA    0.25f

// Dynamic LDS layout (bytes):
//   [0,      131072)  resid  : float[128][256]
//   [131072, 147456)  bstage : __bf16[2][16][256]  (double-buffered code chunks)
//   [147456, 151552)  e2     : float[1024]          (this level's ||e_k||^2)
//   [151552, 152064)  idxs   : int[128]
#define LDS_BYTES   152064
#define OFF_BSTAGE  131072
#define OFF_E2      147456
#define OFF_IDX     151552
#define BSTAGE_ELEMS (16 * DIM)

struct BPair { v8bf a, b; };

__global__ void rvq_zero_commit(float* c) { *c = 0.0f; }

// One wave per code row: e2all[l*K + k] = ||codebooks[l][k]||^2  (4096 rows)
__global__ __launch_bounds__(THREADS)
void rvq_e2_kernel(const float* __restrict__ codebooks, float* __restrict__ e2all)
{
    const int gw   = (int)((blockIdx.x * blockDim.x + threadIdx.x) >> 5);
    const int lane = threadIdx.x & 31;
    if (gw >= MLEV * KCODES) return;
    const float* p = codebooks + (size_t)gw * DIM;
    float s = 0.0f;
    #pragma unroll
    for (int d = lane; d < DIM; d += 32) { float v = p[d]; s += v * v; }
    #pragma unroll
    for (int m = 16; m >= 1; m >>= 1) s += __shfl_xor(s, m, 32);
    if (lane == 0) e2all[gw] = s;
}

__global__ __launch_bounds__(THREADS)
void rvq_kernel(const float* __restrict__ x,
                const float* __restrict__ codebooks,
                const float* __restrict__ e2all,
                float* __restrict__ y,
                float* __restrict__ commit)
{
    extern __shared__ char smem[];
    float*  resid  = (float*)smem;
    __bf16* bstage = (__bf16*)(smem + OFF_BSTAGE);
    float*  e2     = (float*)(smem + OFF_E2);
    int*    idxs   = (int*)(smem + OFF_IDX);

    const int t    = threadIdx.x;
    const int wave = t >> 5;
    const int lane = t & 31;
    const int nloc = lane & 15;          // column / row-in-group selector
    const int hi   = (lane >= 16) ? 1 : 0;
    const long row0 = (long)blockIdx.x * ROWS_WG;

    // Stream residual tile (= x tile) into LDS: thread t owns dim d = t.
    for (int i = 0; i < ROWS_WG; ++i)
        resid[i * DIM + t] = x[(row0 + i) * DIM + t];

    float commit_acc = 0.0f;

    const int rowA = wave * 16 + nloc;   // A-matrix row this lane holds
    const int kbA  = hi ? 16 : 0;        // A k-base (contiguous k = 16h + j, matches B)
    const int kbB  = hi ? 16 : 0;        // B k-base (contiguous k = 16h + j, matches A)

    for (int lvl = 0; lvl < MLEV; ++lvl) {
        const float* cb  = codebooks + (size_t)lvl * KCODES * DIM;
        const float* e2g = e2all + (size_t)lvl * KCODES;

        __syncthreads();  // resid stable; LDS buffers free from previous level

        // Pull this level's precomputed e2 slice into LDS (coalesced, L2-hot).
        #pragma unroll
        for (int c = t; c < KCODES; c += THREADS)
            e2[c] = e2g[c];

        // Build this wave's A fragments: 16 rows x 256 dims in bf16, in regs.
        v16bf A[8];
        {
            const float* rp = resid + rowA * DIM;
            for (int dc = 0; dc < 8; ++dc) {
                v16bf a;
                #pragma unroll
                for (int j = 0; j < 16; ++j) {
                    int k = dc * 32 + kbA + j;
                    a[j] = (__bf16)rp[k];
                }
                A[dc] = a;
            }
        }

        float best[8];
        int   bidx[8];
        #pragma unroll
        for (int v = 0; v < 8; ++v) { best[v] = 3.0e38f; bidx[v] = 0; }

        // Stage a 16-code chunk as bf16: thread t covers dim d = t.
        auto stage = [&](int cc, int buf) {
            const float* src = cb + (size_t)cc * 16 * DIM + t;
            __bf16* dst = bstage + buf * BSTAGE_ELEMS + t;
            #pragma unroll
            for (int i = 0; i < 16; ++i)
                dst[i * DIM] = (__bf16)src[(size_t)i * DIM];
            if (cc + 1 < KCODES / 16)          // warm L2 path for chunk cc+1
                __builtin_prefetch(src + (size_t)16 * DIM, 0, 0);
        };

        stage(0, 0);

        for (int cc = 0; cc < KCODES / 16; ++cc) {
            // One barrier per chunk: makes chunk cc's staging visible AND
            // guarantees buffer (cc+1)&1 (read by WMMA in iter cc-1) is free.
            __syncthreads();

            // Overlap: stage chunk cc+1 while WMMAs consume chunk cc.
            if (cc + 1 < KCODES / 16) stage(cc + 1, (cc + 1) & 1);

            const __bf16* bb = bstage + (cc & 1) * BSTAGE_ELEMS;
            v8f acc = {0.f, 0.f, 0.f, 0.f, 0.f, 0.f, 0.f, 0.f};
            #pragma unroll
            for (int dc = 0; dc < 8; ++dc) {
                const v8bf* bp = (const v8bf*)(bb + nloc * DIM + dc * 32 + kbB);
                BPair pr; pr.a = bp[0]; pr.b = bp[1];
                v16bf B = __builtin_bit_cast(v16bf, pr);
                acc = __builtin_amdgcn_wmma_f32_16x16x32_bf16(
                          false, A[dc], false, B, (short)0, acc, false, false);
            }

            const int   ng  = cc * 16 + nloc;
            const float e2n = e2[ng];
            #pragma unroll
            for (int v = 0; v < 8; ++v) {
                float s = e2n - 2.0f * acc[v];
                if (s < best[v]) { best[v] = s; bidx[v] = ng; }  // keeps lowest idx
            }
        }

        // Cross-lane argmin within each 16-lane half (rows v / v+8).
        #pragma unroll
        for (int v = 0; v < 8; ++v) {
            float bs = best[v]; int bi = bidx[v];
            #pragma unroll
            for (int m = 1; m <= 8; m <<= 1) {
                float os = __shfl_xor(bs, m, 32);
                int   oi = __shfl_xor(bi, m, 32);
                if (os < bs || (os == bs && oi < bi)) { bs = os; bi = oi; }
            }
            best[v] = bs; bidx[v] = bi;
        }
        if (nloc == 0) {
            #pragma unroll
            for (int v = 0; v < 8; ++v)
                idxs[wave * 16 + hi * 8 + v] = bidx[v];
        }
        __syncthreads();

        // Gather winners (L2-hot), update residual, accumulate commit partial:
        // (q - r_l)^2 == r_{l+1}^2
        for (int i = 0; i < ROWS_WG; ++i) {
            int   ci = idxs[i];
            float q  = cb[(size_t)ci * DIM + t];
            float nr = resid[i * DIM + t] - q;
            resid[i * DIM + t] = nr;
            commit_acc += nr * nr;
        }
    }

    __syncthreads();
    // y = sum of quantized vectors = x - final residual (straight-through value)
    for (int i = 0; i < ROWS_WG; ++i)
        y[(row0 + i) * DIM + t] = x[(row0 + i) * DIM + t] - resid[i * DIM + t];

    // Block-reduce commit partials, one atomic per workgroup.
    #pragma unroll
    for (int m = 16; m >= 1; m >>= 1)
        commit_acc += __shfl_xor(commit_acc, m, 32);
    if (lane == 0) e2[wave] = commit_acc;   // e2 region reused as scratch
    __syncthreads();
    if (t == 0) {
        float s = 0.0f;
        for (int w = 0; w < WAVES; ++w) s += e2[w];
        atomicAdd(commit, s * (BETA / ((float)N_TOK * (float)DIM)));
    }
}

extern "C" void kernel_launch(void* const* d_in, const int* in_sizes, int n_in,
                              void* d_out, int out_size, void* d_ws, size_t ws_size,
                              hipStream_t stream) {
    (void)in_sizes; (void)n_in; (void)ws_size; (void)out_size;
    const float* x   = (const float*)d_in[0];
    const float* cbs = (const float*)d_in[1];
    float* y      = (float*)d_out;
    float* commit = (float*)d_out + (size_t)N_TOK * DIM;
    float* e2all  = (float*)d_ws;          // MLEV*KCODES floats = 16 KB scratch

    (void)hipFuncSetAttribute((const void*)rvq_kernel,
                              hipFuncAttributeMaxDynamicSharedMemorySize,
                              LDS_BYTES);

    rvq_zero_commit<<<1, 1, 0, stream>>>(commit);
    rvq_e2_kernel<<<(MLEV * KCODES * 32) / THREADS, THREADS, 0, stream>>>(cbs, e2all);
    rvq_kernel<<<N_TOK / ROWS_WG, THREADS, LDS_BYTES, stream>>>(x, cbs, e2all, y, commit);
}
